// DualGraphEncoder_50757923504168
// MI455X (gfx1250) — compile-verified, matrix operand order checked by reading, not run on hardware
//
#include <hip/hip_runtime.h>
#include <hip/hip_bf16.h>

typedef __bf16 bf16_t;
typedef __attribute__((ext_vector_type(16))) __bf16 v16bf;
typedef __attribute__((ext_vector_type(8)))  float  v8f;

// pointer types for the gfx1250 async-to-LDS builtin (param is an int4-vector
// pointer in global (__device__/AS1) and LDS (__shared__/AS3) address spaces)
typedef int v4i_ __attribute__((__vector_size__(16)));
typedef v4i_ __attribute__((address_space(1)))* gp4i;
typedef v4i_ __attribute__((address_space(3)))* lp4i;

#define BM 128
#define BN 128
#define BK 32
#define LDSK 40   // LDS row pitch in bf16 elems (80B = 20 banks -> conflict-free b128 reads)

#if __has_builtin(__builtin_amdgcn_global_load_async_to_lds_b128) && \
    __has_builtin(__builtin_amdgcn_s_wait_asynccnt)
#define USE_ASYNC_LDS 1
#else
#define USE_ASYNC_LDS 0
#endif

// ---------------------------------------------------------------------------
// Utility kernels
// ---------------------------------------------------------------------------
__global__ void zero_f32_kernel(float* __restrict__ p, size_t n) {
    size_t i = (size_t)blockIdx.x * blockDim.x + threadIdx.x;
    size_t stride = (size_t)gridDim.x * blockDim.x;
    for (; i < n; i += stride) p[i] = 0.0f;
}

__global__ void count_kernel(const long long* __restrict__ dst, int E,
                             float* __restrict__ cnt) {
    int e = blockIdx.x * blockDim.x + threadIdx.x;
    if (e < E) atomicAdd(&cnt[(size_t)dst[e]], 1.0f);
}

// one block per edge; threads stride over features
__global__ void scatter_kernel(const float* __restrict__ x,
                               const long long* __restrict__ src,
                               const long long* __restrict__ dst,
                               int F, float* __restrict__ sum) {
    int e = blockIdx.x;
    long long s = src[e], d = dst[e];
    const float* xs = x + (size_t)s * F;
    float* sd = sum + (size_t)d * F;
    for (int f = threadIdx.x; f < F; f += blockDim.x)
        atomicAdd(&sd[f], xs[f]);
}

// f32 [M,K] -> bf16 [Mpad,Kpad] zero-padded (grid: x over Kpad, y over Mpad rows)
__global__ void cast_pad_kernel(const float* __restrict__ in, bf16_t* __restrict__ out,
                                int M, int K, int Kpad) {
    int i = blockIdx.y;
    int j = blockIdx.x * blockDim.x + threadIdx.x;
    if (j >= Kpad) return;
    float v = (i < M && j < K) ? in[(size_t)i * K + j] : 0.0f;
    out[(size_t)i * Kpad + j] = (bf16_t)v;
}

// sum/max(cnt,1) -> bf16 padded
__global__ void div_cast_kernel(const float* __restrict__ sum,
                                const float* __restrict__ cnt,
                                bf16_t* __restrict__ out,
                                int M, int K, int Kpad) {
    int i = blockIdx.y;
    int j = blockIdx.x * blockDim.x + threadIdx.x;
    if (j >= Kpad) return;
    float v = 0.0f;
    if (i < M && j < K)
        v = sum[(size_t)i * K + j] / fmaxf(cnt[i], 1.0f);
    out[(size_t)i * Kpad + j] = (bf16_t)v;
}

// [R,C] -> [C,R] LDS-tiled transpose, block (32,8)
__global__ void transpose_kernel(const float* __restrict__ in, float* __restrict__ out,
                                 int R, int C) {
    __shared__ float tile[32][33];
    int c0 = blockIdx.x * 32, r0 = blockIdx.y * 32;
    int tx = threadIdx.x, ty = threadIdx.y;
    for (int i = ty; i < 32; i += 8) {
        int r = r0 + i, c = c0 + tx;
        tile[i][tx] = (r < R && c < C) ? in[(size_t)r * C + c] : 0.0f;
    }
    __syncthreads();
    for (int i = ty; i < 32; i += 8) {
        int r = c0 + i, c = r0 + tx;
        if (r < C && c < R) out[(size_t)r * R + c] = tile[tx][i];
    }
}

// ---------------------------------------------------------------------------
// Fused dual GEMM:  D[M,N] = A1 * B1^T + A2 * B2^T + bias
//   A1,A2 : bf16 [Mpad, Kpad] row-major (activations; Mpad = gridDim.y*128)
//   B1,B2 : bf16 [Npad, Kpad] row-major (weights as-is: row n holds K of col n)
//   D     : f32  [M, N]
// 256 threads = 8 waves (wave32). Block tile 128x128; wave tile 32x64 via
// 2x4 grid of 16x16 v_wmma_f32_16x16x32_bf16 (16 WMMA per K-step per wave).
// Staging: double-buffered GLOBAL_LOAD_ASYNC_TO_LDS_B128 (ASYNCcnt pipeline)
// with a peeled, branch-free steady-state loop; sync copy fallback otherwise.
// ---------------------------------------------------------------------------
__global__ __launch_bounds__(256)
void gemm_dual_kernel(const bf16_t* __restrict__ A1, const bf16_t* __restrict__ B1,
                      const bf16_t* __restrict__ A2, const bf16_t* __restrict__ B2,
                      const float* __restrict__ bias, float* __restrict__ D,
                      int M, int N, int Kpad) {
#if USE_ASYNC_LDS
    __shared__ bf16_t lsA1[2][BM * LDSK];
    __shared__ bf16_t lsA2[2][BM * LDSK];
    __shared__ bf16_t lsB1[2][BN * LDSK];
    __shared__ bf16_t lsB2[2][BN * LDSK];
#else
    __shared__ bf16_t lsA1[1][BM * LDSK];
    __shared__ bf16_t lsA2[1][BM * LDSK];
    __shared__ bf16_t lsB1[1][BN * LDSK];
    __shared__ bf16_t lsB2[1][BN * LDSK];
#endif

    const int tid   = threadIdx.x;
    const int lane  = tid & 31;
    const int wave  = tid >> 5;         // 0..7
    const int waveM = wave >> 1;        // 0..3 -> 32-row slab
    const int waveN = wave & 1;         // 0..1 -> 64-col slab
    const int half  = lane >> 4;        // hi/lo 16 lanes
    const int mr    = lane & 15;

    const int mBase = blockIdx.y * BM;
    const int nBase = blockIdx.x * BN;

    v8f acc[2][4] = {};

    const int c0 = half * 8;            // bf16 col offset of fragment chunk 0
    const int c1 = 16 + half * 8;       // chunk 1

    // per-thread staging coordinates: 2 chunks of 16B per matrix per stage
    int srow[2], sq[2], lo[2];
#pragma unroll
    for (int g = 0; g < 2; ++g) {
        int c = tid + g * 256;          // 0..511
        srow[g] = c >> 2;
        sq[g]   = (c & 3) * 8;
        lo[g]   = srow[g] * LDSK + sq[g];
    }

    auto compute = [&](const bf16_t* sA1, const bf16_t* sA2,
                       const bf16_t* sB1, const bf16_t* sB2) {
        v16bf a1[2], a2[2], b1[4], b2[4];
#pragma unroll
        for (int am = 0; am < 2; ++am) {
            const bf16_t* ba = sA1 + (waveM * 32 + am * 16 + mr) * LDSK;
            const bf16_t* bb = sA2 + (waveM * 32 + am * 16 + mr) * LDSK;
            ((uint4*)&a1[am])[0] = *(const uint4*)(ba + c0);
            ((uint4*)&a1[am])[1] = *(const uint4*)(ba + c1);
            ((uint4*)&a2[am])[0] = *(const uint4*)(bb + c0);
            ((uint4*)&a2[am])[1] = *(const uint4*)(bb + c1);
        }
#pragma unroll
        for (int bn = 0; bn < 4; ++bn) {
            const bf16_t* ba = sB1 + (waveN * 64 + bn * 16 + mr) * LDSK;
            const bf16_t* bb = sB2 + (waveN * 64 + bn * 16 + mr) * LDSK;
            ((uint4*)&b1[bn])[0] = *(const uint4*)(ba + c0);
            ((uint4*)&b1[bn])[1] = *(const uint4*)(ba + c1);
            ((uint4*)&b2[bn])[0] = *(const uint4*)(bb + c0);
            ((uint4*)&b2[bn])[1] = *(const uint4*)(bb + c1);
        }
#pragma unroll
        for (int am = 0; am < 2; ++am)
#pragma unroll
            for (int bn = 0; bn < 4; ++bn) {
                acc[am][bn] = __builtin_amdgcn_wmma_f32_16x16x32_bf16(
                    false, a1[am], false, b1[bn], (short)0, acc[am][bn], false, false);
                acc[am][bn] = __builtin_amdgcn_wmma_f32_16x16x32_bf16(
                    false, a2[am], false, b2[bn], (short)0, acc[am][bn], false, false);
            }
    };

#if USE_ASYNC_LDS
    // running per-thread global pointers (advance by BK elems per stage)
    const bf16_t* gA1[2]; const bf16_t* gA2[2];
    const bf16_t* gB1[2]; const bf16_t* gB2[2];
#pragma unroll
    for (int g = 0; g < 2; ++g) {
        size_t oA = (size_t)(mBase + srow[g]) * Kpad + sq[g];
        size_t oB = (size_t)(nBase + srow[g]) * Kpad + sq[g];
        gA1[g] = A1 + oA; gA2[g] = A2 + oA;
        gB1[g] = B1 + oB; gB2[g] = B2 + oB;
    }

    // issue one stage (8 async b128 per thread; LDS written directly, no VGPRs)
    auto issue = [&](int s) {
#pragma unroll
        for (int g = 0; g < 2; ++g) {
            __builtin_amdgcn_global_load_async_to_lds_b128(
                (gp4i)gA1[g], (lp4i)(&lsA1[s][lo[g]]), 0, 0);
            __builtin_amdgcn_global_load_async_to_lds_b128(
                (gp4i)gA2[g], (lp4i)(&lsA2[s][lo[g]]), 0, 0);
            __builtin_amdgcn_global_load_async_to_lds_b128(
                (gp4i)gB1[g], (lp4i)(&lsB1[s][lo[g]]), 0, 0);
            __builtin_amdgcn_global_load_async_to_lds_b128(
                (gp4i)gB2[g], (lp4i)(&lsB2[s][lo[g]]), 0, 0);
            gA1[g] += BK; gA2[g] += BK; gB1[g] += BK; gB2[g] += BK;
        }
    };

    const int KT = Kpad / BK;           // >= 2 for all our shapes
    issue(0);
    // branch-free steady state: prefetch kt+1 while computing kt
    for (int kt = 0; kt < KT - 1; ++kt) {
        const int cur = kt & 1;
        __syncthreads();                       // all waves done reading buf[cur^1]
        issue(cur ^ 1);                        // overlap next panel with compute
        __builtin_amdgcn_s_wait_asynccnt(8);   // stage kt landed (in-order)
        __syncthreads();                       // stage kt visible workgroup-wide
        compute(lsA1[cur], lsA2[cur], lsB1[cur], lsB2[cur]);
    }
    // epilogue: last panel
    {
        const int cur = (KT - 1) & 1;
        __builtin_amdgcn_s_wait_asynccnt(0);
        __syncthreads();
        compute(lsA1[cur], lsA2[cur], lsB1[cur], lsB2[cur]);
    }
#else
    for (int kb = 0; kb < Kpad; kb += BK) {
        __syncthreads();                // previous compute done before overwrite
#pragma unroll
        for (int g = 0; g < 2; ++g) {
            const int row = srow[g], q = sq[g];
            size_t gA = (size_t)(mBase + row) * Kpad + kb + q;
            size_t gB = (size_t)(nBase + row) * Kpad + kb + q;
            *(uint4*)(&lsA1[0][row * LDSK + q]) = *(const uint4*)(A1 + gA);
            *(uint4*)(&lsA2[0][row * LDSK + q]) = *(const uint4*)(A2 + gA);
            *(uint4*)(&lsB1[0][row * LDSK + q]) = *(const uint4*)(B1 + gB);
            *(uint4*)(&lsB2[0][row * LDSK + q]) = *(const uint4*)(B2 + gB);
            if (kb + BK < Kpad) {
                __builtin_prefetch(A1 + gA + BK, 0, 0);
                __builtin_prefetch(B1 + gB + BK, 0, 0);
            }
        }
        __syncthreads();
        compute(lsA1[0], lsA2[0], lsB1[0], lsB2[0]);
    }
#endif

    // store D = acc + bias (C/D layout: VGPR r -> m = r + 8*half, n = lane&15)
#pragma unroll
    for (int bn = 0; bn < 4; ++bn) {
        int col = nBase + waveN * 64 + bn * 16 + mr;
        float bv = (col < N) ? bias[col] : 0.0f;
#pragma unroll
        for (int am = 0; am < 2; ++am) {
#pragma unroll
            for (int r = 0; r < 8; ++r) {
                int row = mBase + waveM * 32 + am * 16 + half * 8 + r;
                if (row < M && col < N)
                    D[(size_t)row * N + col] = acc[am][bn][r] + bv;
            }
        }
    }
}

// ---------------------------------------------------------------------------
// Host orchestration
// ---------------------------------------------------------------------------
extern "C" void kernel_launch(void* const* d_in, const int* in_sizes, int n_in,
                              void* d_out, int out_size, void* d_ws, size_t ws_size,
                              hipStream_t stream) {
    (void)in_sizes; (void)n_in; (void)out_size; (void)ws_size;

    enum { NC = 10000, NG = 2000, EC = 320000, EG = 64000 };
    const int KpadC = 2048,  MpadC = 10112;   // cell phase: M=10000 nodes, K=N=2000
    const int KpadG = 10016, MpadG = 2048;    // gene phase: M=2000 nodes,  K=N=10000
    const int NpadC = 2048,  NpadG = 10112;   // weight row padding (output features)

    const float*     x   = (const float*)d_in[0];
    const long long* cg  = (const long long*)d_in[1];
    const long long* gg  = (const long long*)d_in[2];
    const float*     WlC = (const float*)d_in[3];
    const float*     WrC = (const float*)d_in[4];
    const float*     bC  = (const float*)d_in[5];
    const float*     WlG = (const float*)d_in[6];
    const float*     WrG = (const float*)d_in[7];
    const float*     bG  = (const float*)d_in[8];
    float*           out = (float*)d_out;

    // workspace carve (256B aligned)
    char* wp = (char*)d_ws;
    auto carve = [&](size_t bytes) -> void* {
        void* r = (void*)wp;
        wp += (bytes + 255) & ~(size_t)255;
        return r;
    };
    float*  bufA = (float*)carve(sizeof(float) * (size_t)NC * NG);
    float*  bufS = (float*)carve(sizeof(float) * (size_t)NC * NG);
    float*  cnt  = (float*)carve(sizeof(float) * NC);
    bf16_t* xbf  = (bf16_t*)carve(2ull * MpadC * KpadC);   // max of both phases
    bf16_t* agf  = (bf16_t*)carve(2ull * MpadC * KpadC);
    bf16_t* wlc  = (bf16_t*)carve(2ull * NpadC * KpadC);
    bf16_t* wrc  = (bf16_t*)carve(2ull * NpadC * KpadC);
    bf16_t* wlg  = (bf16_t*)carve(2ull * NpadG * KpadG);
    bf16_t* wrg  = (bf16_t*)carve(2ull * NpadG * KpadG);

    // one-time weight conversion to bf16 ([N,K] kept as-is == WMMA B layout [n][k])
    cast_pad_kernel<<<dim3((KpadC + 255) / 256, NpadC), 256, 0, stream>>>(WlC, wlc, NG, NG, KpadC);
    cast_pad_kernel<<<dim3((KpadC + 255) / 256, NpadC), 256, 0, stream>>>(WrC, wrc, NG, NG, KpadC);
    cast_pad_kernel<<<dim3((KpadG + 255) / 256, NpadG), 256, 0, stream>>>(WlG, wlg, NC, NC, KpadG);
    cast_pad_kernel<<<dim3((KpadG + 255) / 256, NpadG), 256, 0, stream>>>(WrG, wrg, NC, NC, KpadG);

    auto conv = [&](const float* cur, float* sum, int nodes, int F, int Mpad, int Kpad,
                    const long long* esrc, const long long* edst, int E,
                    const bf16_t* wl, const bf16_t* wr, const float* bias, float* dstF) {
        zero_f32_kernel<<<dim3(2048), 256, 0, stream>>>(sum, (size_t)nodes * F);
        zero_f32_kernel<<<dim3(40),   256, 0, stream>>>(cnt, (size_t)nodes);
        count_kernel<<<dim3((E + 255) / 256), 256, 0, stream>>>(edst, E, cnt);
        scatter_kernel<<<dim3(E), 256, 0, stream>>>(cur, esrc, edst, F, sum);
        cast_pad_kernel<<<dim3((Kpad + 255) / 256, Mpad), 256, 0, stream>>>(cur, xbf, nodes, F, Kpad);
        div_cast_kernel<<<dim3((Kpad + 255) / 256, Mpad), 256, 0, stream>>>(sum, cnt, agf, nodes, F, Kpad);
        gemm_dual_kernel<<<dim3((F + 127) / 128, Mpad / 128), 256, 0, stream>>>(
            agf, wl, xbf, wr, bias, dstF, nodes, F, Kpad);
    };

    // 3x cell-graph SAGE conv on x[10000,2000] (shared weights across timepoints)
    conv(x,    bufS, NC, NG, MpadC, KpadC, cg, cg + EC, EC, wlc, wrc, bC, bufA);
    conv(bufA, bufS, NC, NG, MpadC, KpadC, cg, cg + EC, EC, wlc, wrc, bC, bufA);
    conv(bufA, bufS, NC, NG, MpadC, KpadC, cg, cg + EC, EC, wlc, wrc, bC, bufA);

    // x = x.T  -> [2000, 10000]
    transpose_kernel<<<dim3((NG + 31) / 32, (NC + 31) / 32), dim3(32, 8), 0, stream>>>(
        bufA, bufS, NC, NG);

    // 3x gene-graph SAGE conv on [2000,10000]; final layer writes d_out
    conv(bufS, bufA, NG, NC, MpadG, KpadG, gg, gg + EG, EG, wlg, wrg, bG, bufS);
    conv(bufS, bufA, NG, NC, MpadG, KpadG, gg, gg + EG, EG, wlg, wrg, bG, bufS);
    conv(bufS, bufA, NG, NC, MpadG, KpadG, gg, gg + EG, EG, wlg, wrg, bG, out);
}
